// ContrastiveLoss_31568009626146
// MI455X (gfx1250) — matrix operand order repaired and checked by reading, not compiled, not run on hardware
//
#include <hip/hip_runtime.h>
#include <hip/hip_bf16.h>

// ---------------------------------------------------------------------------
// NT-Xent contrastive loss, fused: never materialize the 8192x8192 sim matrix
// (saves ~512MB HBM traffic @23.3TB/s; inputs are 4MB and L2-resident).
// GEMM on v_wmma_f32_16x16x32_f16; B panels staged with double-buffered
// global_load_async_to_lds_b128 (ASYNCcnt) so DMA overlaps WMMA compute.
// ---------------------------------------------------------------------------

typedef __attribute__((ext_vector_type(16))) _Float16 v16h;
typedef __attribute__((ext_vector_type(8)))  _Float16 v8h;
typedef __attribute__((ext_vector_type(4)))  _Float16 v4h;
typedef __attribute__((ext_vector_type(8)))  float    v8f;

#define D_DIM 128
#define INV_TAU 10.0f
// exp(s/tau) == exp2(s * INV_TAU * log2(e)); fold both constants into one mul
#define EXP2_SCALE (INV_TAU * 1.4426950408889634f)
// raw v_exp_f32 (inputs bounded by +-INV_TAU*log2e, no denorm concerns)
#define FAST_EXP2(x) __builtin_amdgcn_exp2f(x)

// ---------------- Kernel 1: normalize rows, emit f16 rn ----------------
// reps = concat([zjs, zis]); one wave (32 lanes) per row, 4 floats/lane.
__global__ __launch_bounds__(256) void nrm_kernel(const float* __restrict__ zis,
                                                  const float* __restrict__ zjs,
                                                  _Float16* __restrict__ rn,
                                                  int Bv) {
  const int wave = threadIdx.x >> 5;
  const int lane = threadIdx.x & 31;
  const int row  = blockIdx.x * 8 + wave;     // 8 waves per block
  const float* src = (row < Bv) ? (zjs + (size_t)row * D_DIM)
                                : (zis + (size_t)(row - Bv) * D_DIM);
  float4 x = ((const float4*)src)[lane];
  float ss = x.x * x.x + x.y * x.y + x.z * x.z + x.w * x.w;
#pragma unroll
  for (int m = 16; m >= 1; m >>= 1) ss += __shfl_xor(ss, m, 32);
  float inv = 1.0f / fmaxf(sqrtf(ss), 1e-8f);
  v4h o;
  o[0] = (_Float16)(x.x * inv); o[1] = (_Float16)(x.y * inv);
  o[2] = (_Float16)(x.z * inv); o[3] = (_Float16)(x.w * inv);
  *(v4h*)(rn + (size_t)row * D_DIM + 4 * lane) = o;
}

// ---------------- Kernel 2: fused Gram-GEMM + row softmax stats ----------------
// Grid: N/128 workgroups of 256 threads (8 waves).  Wave w owns rows
// [blk*128 + 16w, +16); A fragments register-resident.  128-row column panels
// (32KB) double-buffered in LDS via async-to-LDS DMA.
__global__ __launch_bounds__(256) void sim_kernel(const _Float16* __restrict__ rn,
                                                  float* __restrict__ pos,
                                                  float* __restrict__ esum,
                                                  int N) {
  __shared__ __align__(32) _Float16 panel[2][128 * D_DIM];   // 2 x 32 KB

  const int tid  = threadIdx.x;
  const int wave = tid >> 5;
  const int lane = tid & 31;
  const int n    = lane & 15;     // output column within tile / A row id
  const int hi   = lane >> 4;     // half-wave select
  const int row_base = blockIdx.x * 128 + wave * 16;

  // --- A fragments: 16 rows x 128 K, contiguous k order (same as B).
  // lane (n,hi): row = row_base+n; K halfs: [32ks+16hi,+16) contiguous
  const _Float16* arow = rn + (size_t)(row_base + n) * D_DIM;
  v16h afrag[4];
#pragma unroll
  for (int ks = 0; ks < 4; ++ks) {
    afrag[ks] = *(const v16h*)(arow + 32 * ks + 16 * hi);
  }

  float expAcc[8];
  float posAcc[8];
#pragma unroll
  for (int r = 0; r < 8; ++r) { expAcc[r] = 0.0f; posAcc[r] = 0.0f; }

  // --- async panel stage: 32KB = 128B/thread = 8 x b128, VGPR-free DMA.
  // ISA: INST_OFFSET is added to BOTH the LDS dest and the global address,
  // so one (vdst, vaddr) pair + 8 literal offsets covers the panel.
  const unsigned lds0 = (unsigned)(size_t)(&panel[0][0]) + (unsigned)tid * 16u;
  auto stage = [&](int p, int buf) {
    unsigned ldsd = lds0 + (unsigned)buf * 32768u;
    unsigned goff = (unsigned)p * 32768u + (unsigned)tid * 16u;
#define A16(OFF)                                                            \
    asm volatile("global_load_async_to_lds_b128 %0, %1, %2 offset:" #OFF    \
                 :: "v"(ldsd), "v"(goff), "s"(rn) : "memory")
    A16(0); A16(4096); A16(8192); A16(12288);
    A16(16384); A16(20480); A16(24576); A16(28672);
#undef A16
  };

  // --- one 16x16 output tile: preload 4 B frags, then 4 chained WMMAs.
  auto tile = [&](const _Float16* pan, int ct) -> v8f {
    v16h bf[4];
#pragma unroll
    for (int ks = 0; ks < 4; ++ks) {
      // B (KxN) frags, same K order as A: lanes 0-15 hold K=kb..kb+15 contiguous, lanes 16-31 +16
      bf[ks] = *(const v16h*)(pan + (size_t)(16 * ct + n) * D_DIM +
                              32 * ks + 16 * hi);
    }
    v8f acc = {};
#pragma unroll
    for (int ks = 0; ks < 4; ++ks) {
      acc = __builtin_amdgcn_wmma_f32_16x16x32_f16(
          false, afrag[ks], false, bf[ks], (short)0, acc, false, false);
    }
    return acc;
  };

  const int NP = N / 128;                 // 64 panels
  const int self_p = blockIdx.x;          // panel holding this block's diagonal
  const int part_p = self_p ^ (NP >> 1);  // panel holding the positives

  stage(0, 0);
  asm volatile("s_wait_asynccnt 0x0" ::: "memory");
  __syncthreads();

  for (int p = 0; p < NP; ++p) {
    const int buf = p & 1;
    if (p + 1 < NP) stage(p + 1, buf ^ 1);   // fire-and-forget next panel DMA
    const _Float16* pan = panel[buf];

    if (p == self_p || p == part_p) {
      // rare path (2/64 panels): handle diagonal + positive bookkeeping
      const int c0 = p * 128;
#pragma unroll 2
      for (int ct = 0; ct < 8; ++ct) {
        v8f acc = tile(pan, ct);
        const int j = c0 + 16 * ct + n;
#pragma unroll
        for (int r = 0; r < 8; ++r) {
          const int i = row_base + r + 8 * hi;   // C layout: M = r + 8*hi
          float s = acc[r];
          if (j != i) expAcc[r] += FAST_EXP2(s * EXP2_SCALE);
          if (j == (i ^ (N >> 1))) posAcc[r] = s;
        }
      }
    } else {
      // hot path: pure accumulate, single fused scale into the exp2 unit
#pragma unroll 2
      for (int ct = 0; ct < 8; ++ct) {
        v8f acc = tile(pan, ct);
#pragma unroll
        for (int r = 0; r < 8; ++r) expAcc[r] += FAST_EXP2(acc[r] * EXP2_SCALE);
      }
    }

    if (p + 1 < NP) {
      asm volatile("s_wait_asynccnt 0x0" ::: "memory");  // own DMA landed
      __syncthreads();                                   // everyone's DMA landed
    }
  }

  // reduce across the 16 lanes (n = 0..15) that share each row
#pragma unroll
  for (int r = 0; r < 8; ++r) {
#pragma unroll
    for (int m = 1; m <= 8; m <<= 1) {
      expAcc[r] += __shfl_xor(expAcc[r], m, 32);
      posAcc[r] += __shfl_xor(posAcc[r], m, 32);   // non-writers hold exact 0
    }
  }
  if (n == 0) {   // lanes 0 (rows m=0..7) and 16 (rows m=8..15)
#pragma unroll
    for (int r = 0; r < 8; ++r) {
      const int i = row_base + r + 8 * hi;
      esum[i] = expAcc[r];
      pos[i]  = posAcc[r];
    }
  }
}

// ---------------- Kernel 3: per-row loss terms + block reduction ----------------
__global__ __launch_bounds__(256) void rowred_kernel(const float* __restrict__ pos,
                                                     const float* __restrict__ esum,
                                                     float* __restrict__ blkCE,
                                                     float* __restrict__ blkPT,
                                                     int N) {
  __shared__ float sc[8], sp[8];
  const int i    = blockIdx.x * 256 + threadIdx.x;
  const int wave = threadIdx.x >> 5;
  const int lane = threadIdx.x & 31;
  float ce = 0.0f, pt = 0.0f;
  if (i < N) {
    float p   = pos[i] * INV_TAU;
    float lse = __logf(esum[i]);   // esum holds sum exp(sim/tau) exactly
    ce = lse - p;                  // -log_softmax[0]
    pt = __expf(p - lse);          // softmax prob of the positive
  }
#pragma unroll
  for (int m = 16; m >= 1; m >>= 1) {
    ce += __shfl_xor(ce, m, 32);
    pt += __shfl_xor(pt, m, 32);
  }
  if (lane == 0) { sc[wave] = ce; sp[wave] = pt; }
  __syncthreads();
  if (threadIdx.x == 0) {
    float tce = 0.0f, tpt = 0.0f;
#pragma unroll
    for (int w = 0; w < 8; ++w) { tce += sc[w]; tpt += sp[w]; }
    blkCE[blockIdx.x] = tce;
    blkPT[blockIdx.x] = tpt;
  }
}

// ---------------- Kernel 4: final scalar ----------------
__global__ void fin_kernel(const float* __restrict__ blkCE,
                           const float* __restrict__ blkPT,
                           float* __restrict__ out, int nblk, int N, int Bv) {
  const int lane = threadIdx.x;   // 32 threads, nblk == 32
  float ce = (lane < nblk) ? blkCE[lane] : 0.0f;
  float pt = (lane < nblk) ? blkPT[lane] : 0.0f;
#pragma unroll
  for (int m = 16; m >= 1; m >>= 1) {
    ce += __shfl_xor(ce, m, 32);
    pt += __shfl_xor(pt, m, 32);
  }
  if (lane == 0) {
    // loss = CE/N + B*(1/B - ptSum/(N*(N-1))) = CE/N + 1 - B*pt
    float ptm = pt / ((float)N * (float)(N - 1));
    out[0] = ce / (float)N + 1.0f - (float)Bv * ptm;
  }
}

extern "C" void kernel_launch(void* const* d_in, const int* in_sizes, int n_in,
                              void* d_out, int out_size, void* d_ws, size_t ws_size,
                              hipStream_t stream) {
  const float* zis = (const float*)d_in[0];
  const float* zjs = (const float*)d_in[1];
  float* out = (float*)d_out;

  const int Bv = in_sizes[0] / D_DIM;   // 4096
  const int N  = 2 * Bv;                // 8192

  // workspace layout
  char* ws = (char*)d_ws;
  _Float16* rn  = (_Float16*)ws;                       // N*D f16  (2 MB)
  size_t off = (size_t)N * D_DIM * sizeof(_Float16);
  float* pos   = (float*)(ws + off);  off += (size_t)N * sizeof(float);
  float* esum  = (float*)(ws + off);  off += (size_t)N * sizeof(float);
  float* blkCE = (float*)(ws + off);  off += 256 * sizeof(float);
  float* blkPT = (float*)(ws + off);

  nrm_kernel<<<N / 8, 256, 0, stream>>>(zis, zjs, rn, Bv);
  sim_kernel<<<N / 128, 256, 0, stream>>>(rn, pos, esum, N);
  const int nblk = N / 256;
  rowred_kernel<<<nblk, 256, 0, stream>>>(pos, esum, blkCE, blkPT, N);
  fin_kernel<<<1, 32, 0, stream>>>(blkCE, blkPT, out, nblk, N, Bv);
}
